// Decoder_55078660603927
// MI455X (gfx1250) — matrix operand order repaired and checked by reading, not compiled, not run on hardware
//
#include <hip/hip_runtime.h>
#include <stdint.h>

typedef __attribute__((ext_vector_type(16))) __bf16       v16bf;
typedef __attribute__((ext_vector_type(8)))  float        v8f;

#define BATCH 16384
#define H     256
#define NOUT  31
#define STEPS 120
#define ROW_STRIDE 268   // 12-float skew -> conflict-free ds b128 A-builds

// ---- workspace layout (uint16 element offsets for weights) ----
#define WRZ_OFF    0        // combined (W_ih+W_hh) rz: 32 ntiles x 8 kt = 256 frags
#define WHHRZ_OFF  131072   // W_hh rz (step 0): 256 frags
#define WIN_OFF    262144   // W_ih n-gate: 128 frags
#define WHN_OFF    327680   // W_hh n-gate: 128 frags
#define WOUT_OFF   393216   // W_out padded to 32 cols: 16 frags
#define WZERO_OFF  401408   // zero weights (step-0 i_n): 128 frags
#define TOT_W_ELEMS (912 * 512)
#define BIAS_BYTE_OFF (TOT_W_ELEMS * 2)   // brz[512] bin[256] bhn[256] bout[32]

__device__ __forceinline__ uint16_t f2bf_bits(float f) {   // prep-kernel use (RNE)
  uint32_t u = __float_as_uint(f);
  uint32_t r = u + 0x7FFFu + ((u >> 16) & 1u);
  return (uint16_t)(r >> 16);
}

__device__ __forceinline__ float fast_rcp(float x) {
#if __has_builtin(__builtin_amdgcn_rcpf)
  return __builtin_amdgcn_rcpf(x);
#else
  return 1.f / x;
#endif
}
__device__ __forceinline__ float fast_sigmoid(float x) {
  return fast_rcp(1.f + __expf(-x));
}
__device__ __forceinline__ float fast_tanh(float x) {
#if __has_builtin(__builtin_amdgcn_tanhf)
  return __builtin_amdgcn_tanhf(x);
#else
  x = fminf(fmaxf(x, -12.f), 12.f);
  float e2 = __expf(2.f * x);
  return (e2 - 1.f) * fast_rcp(e2 + 1.f);
#endif
}

__device__ __forceinline__ v8f wmma_acc(v16bf a, v16bf b, v8f c) {
  return __builtin_amdgcn_wmma_f32_16x16x32_bf16(false, a, false, b, (short)0, c, false, false);
}

// Build one 16x32 bf16 A-fragment from fp32 LDS state (CDNA5 16-bit A layout).
// Forced float4 reads -> ds_load_b128; native __bf16 casts -> hw cvt (packable).
__device__ __forceinline__ v16bf buildA(const float* __restrict__ s) {
  float4 a0 = *(const float4*)(s);        // K 0..3  (per-lane half)
  float4 a1 = *(const float4*)(s + 4);    // K 4..7
  float4 b0 = *(const float4*)(s + 16);   // K 16..19
  float4 b1 = *(const float4*)(s + 20);   // K 20..23
  v16bf r;
  r[0]  = (__bf16)a0.x; r[1]  = (__bf16)a0.y; r[2]  = (__bf16)a0.z; r[3]  = (__bf16)a0.w;
  r[4]  = (__bf16)a1.x; r[5]  = (__bf16)a1.y; r[6]  = (__bf16)a1.z; r[7]  = (__bf16)a1.w;
  r[8]  = (__bf16)b0.x; r[9]  = (__bf16)b0.y; r[10] = (__bf16)b0.z; r[11] = (__bf16)b0.w;
  r[12] = (__bf16)b1.x; r[13] = (__bf16)b1.y; r[14] = (__bf16)b1.z; r[15] = (__bf16)b1.w;
  return r;
}

// One 16-load clause: all 8 B frags of a gate chain (pre-swizzled, lane*32B coalesced)
__device__ __forceinline__ void loadB8(v16bf B[8], const uint16_t* __restrict__ base,
                                       int frag0, int lane) {
#pragma unroll
  for (int f = 0; f < 8; ++f)
    B[f] = *(const v16bf*)(base + (size_t)(frag0 + f) * 512 + lane * 16);
}

// 16-WMMA chain: two 16-row M-tiles share every B fragment (32 flops/byte from L2)
__device__ __forceinline__ void chain2(v8f& a0, v8f& a1, const v16bf A0[8],
                                       const v16bf A1[8], const v16bf B[8]) {
#pragma unroll
  for (int kt = 0; kt < 8; ++kt) {
    a0 = wmma_acc(A0[kt], B[kt], a0);
    a1 = wmma_acc(A1[kt], B[kt], a1);
  }
}

// Gates with biases already folded into the WMMA C-operand.
__device__ __forceinline__ void gate_update(const v8f& R, const v8f& Z, const v8f& I,
                                            const v8f& Hn, float* __restrict__ hcol) {
#pragma unroll
  for (int v = 0; v < 8; ++v) {
    float r = fast_sigmoid(R[v]);
    float z = fast_sigmoid(Z[v]);
    float n = fast_tanh(I[v] + r * Hn[v]);
    float hold = hcol[v * ROW_STRIDE];
    hcol[v * ROW_STRIDE] = n + z * (hold - n);   // (1-z)*n + z*h
  }
}

// ---------------- prep kernels ----------------
__global__ void prep_weights(const float* __restrict__ Wih, const float* __restrict__ Whh,
                             const float* __restrict__ Wo, uint16_t* __restrict__ wbase) {
  int tid = blockIdx.x * blockDim.x + threadIdx.x;
  if (tid >= TOT_W_ELEMS) return;
  int frag = tid >> 9;
  int idx  = tid & 511;
  int L = idx >> 4, e = (idx & 7) | ((idx & 8) << 1);
  int nl = L & 15, khalf = (L >> 4) * 8;

  uint16_t val = 0;
  uint16_t* dst;
  int f;
  if (frag < 256) {
    f = frag;
    int n = (f >> 3) * 16 + nl, k = (f & 7) * 32 + khalf + e;
    val = f2bf_bits(Wih[n * H + k] + Whh[n * H + k]);
    dst = wbase + WRZ_OFF;
  } else if (frag < 512) {
    f = frag - 256;
    int n = (f >> 3) * 16 + nl, k = (f & 7) * 32 + khalf + e;
    val = f2bf_bits(Whh[n * H + k]);
    dst = wbase + WHHRZ_OFF;
  } else if (frag < 640) {
    f = frag - 512;
    int n = (f >> 3) * 16 + nl, k = (f & 7) * 32 + khalf + e;
    val = f2bf_bits(Wih[(512 + n) * H + k]);
    dst = wbase + WIN_OFF;
  } else if (frag < 768) {
    f = frag - 640;
    int n = (f >> 3) * 16 + nl, k = (f & 7) * 32 + khalf + e;
    val = f2bf_bits(Whh[(512 + n) * H + k]);
    dst = wbase + WHN_OFF;
  } else if (frag < 784) {
    f = frag - 768;
    int n = (f >> 3) * 16 + nl, k = (f & 7) * 32 + khalf + e;
    val = (n < NOUT) ? f2bf_bits(Wo[n * H + k]) : (uint16_t)0;
    dst = wbase + WOUT_OFF;
  } else {
    f = frag - 784;
    val = 0;
    dst = wbase + WZERO_OFF;
  }
  dst[f * 512 + idx] = val;
}

__global__ void prep_bias(const float* __restrict__ bih, const float* __restrict__ bhh,
                          const float* __restrict__ bo, float* __restrict__ bb) {
  int i = blockIdx.x * blockDim.x + threadIdx.x;
  if (i < 512)       bb[i] = bih[i] + bhh[i];
  else if (i < 768)  bb[i] = bih[i];                    // b_in = b_ih[512..767]
  else if (i < 1024) bb[i] = bhh[512 + (i - 768)];      // b_hn
  else if (i < 1056) { int o = i - 1024; bb[i] = (o < NOUT) ? bo[o] : 0.f; }
}

// ---------------- main: one wave = 32 batch rows, 120 sequential steps ----------------
__global__ __launch_bounds__(32) void gru_decoder_main(
    const float* __restrict__ h0, const uint16_t* __restrict__ wbase,
    const float* __restrict__ bbase, float* __restrict__ out) {
  __shared__ float hs[32 * ROW_STRIDE];   // fp32 state, 32 rows, skewed
  __shared__ float sb[1056];

  const int lane = threadIdx.x & 31;
  const int wrow = blockIdx.x * 32;

  for (int i = lane; i < 1056; i += 32) sb[i] = bbase[i];
  for (int i = lane; i < 32 * (H / 4); i += 32) {
    int r = i >> 6, c4 = i & 63;
    float4 val = ((const float4*)(h0 + (size_t)(wrow + r) * H))[c4];
    *(float4*)&hs[r * ROW_STRIDE + c4 * 4] = val;
  }
  __syncthreads();

  const uint16_t* Wrz   = wbase + WRZ_OFF;
  const uint16_t* Whhrz = wbase + WHHRZ_OFF;
  const uint16_t* Win   = wbase + WIN_OFF;
  const uint16_t* Whn   = wbase + WHN_OFF;
  const uint16_t* Wout  = wbase + WOUT_OFF;
  const uint16_t* Wzero = wbase + WZERO_OFF;
  const float* sbrz = sb;
  const float* sbin = sb + 512;
  const float* sbhn = sb + 768;
  const float* sbout = sb + 1024;

  const int row  = lane & 15;
  const int koff = (lane >> 4) * 8;
  const int ncol = lane & 15;
  const int m0   = (lane >> 4) * 8;
  const float* arow0 = hs + row * ROW_STRIDE + koff;         // rows 0..15
  const float* arow1 = hs + (16 + row) * ROW_STRIDE + koff;  // rows 16..31

  v16bf A0[8], A1[8];
#pragma unroll
  for (int f = 0; f < 8; ++f) { A0[f] = buildA(arow0 + f * 32); A1[f] = buildA(arow1 + f * 32); }

#pragma unroll 1
  for (int t = 0; t < STEPS; ++t) {
    const uint16_t* WrzT = (t == 0) ? Whhrz : Wrz;   // step 0: x=0 -> rz from W_hh only
    const uint16_t* WinT = (t == 0) ? Wzero : Win;   // step 0: i_n matmul contributes 0

#pragma unroll 1
    for (int jt = 0; jt < 16; ++jt) {
      const int c = jt * 16 + ncol;
      const float br = sbrz[c], bz = sbrz[256 + c];
      const float bi = sbin[c], bh = sbhn[c];
      // biases ride in the WMMA C-operand
      v8f accR0 = {br,br,br,br,br,br,br,br}, accR1 = accR0;
      v8f accZ0 = {bz,bz,bz,bz,bz,bz,bz,bz}, accZ1 = accZ0;
      v8f accI0 = {bi,bi,bi,bi,bi,bi,bi,bi}, accI1 = accI0;
      v8f accH0 = {bh,bh,bh,bh,bh,bh,bh,bh}, accH1 = accH0;

      v16bf B[8];
      loadB8(B, WrzT, jt * 8, lane);                                  // R
      __builtin_prefetch(WrzT + (size_t)(16 + jt) * 8 * 512 + lane * 16, 0, 3);
      chain2(accR0, accR1, A0, A1, B);
      loadB8(B, WrzT, (16 + jt) * 8, lane);                           // Z
      __builtin_prefetch(WinT + (size_t)jt * 8 * 512 + lane * 16, 0, 3);
      chain2(accZ0, accZ1, A0, A1, B);
      loadB8(B, WinT, jt * 8, lane);                                  // i_n
      __builtin_prefetch(Whn + (size_t)jt * 8 * 512 + lane * 16, 0, 3);
      chain2(accI0, accI1, A0, A1, B);
      loadB8(B, Whn, jt * 8, lane);                                   // h_n
      __builtin_prefetch(WrzT + (size_t)((jt + 1) & 15) * 8 * 512 + lane * 16, 0, 3);
      chain2(accH0, accH1, A0, A1, B);

      gate_update(accR0, accZ0, accI0, accH0, hs + m0 * ROW_STRIDE + c);
      gate_update(accR1, accZ1, accI1, accH1, hs + (16 + m0) * ROW_STRIDE + c);
    }

    // refresh A from h_{t+1} (used by output proj and next step)
#pragma unroll
    for (int f = 0; f < 8; ++f) { A0[f] = buildA(arow0 + f * 32); A1[f] = buildA(arow1 + f * 32); }

    // y_t = h_{t+1} @ W_out^T + b_out  (31 cols padded to 32)
#pragma unroll
    for (int ot = 0; ot < 2; ++ot) {
      const int o = ot * 16 + ncol;
      const float bo = sbout[o];            // padded to 32 -> safe
      v16bf B[8];
      loadB8(B, Wout, ot * 8, lane);
      v8f o0 = {bo,bo,bo,bo,bo,bo,bo,bo}, o1 = o0;
      chain2(o0, o1, A0, A1, B);
      if (o < NOUT) {
#pragma unroll
        for (int v = 0; v < 8; ++v) {
          out[(size_t)(wrow + m0 + v) * (STEPS * NOUT) + (size_t)t * NOUT + o] = o0[v];
          out[(size_t)(wrow + 16 + m0 + v) * (STEPS * NOUT) + (size_t)t * NOUT + o] = o1[v];
        }
      }
    }
  }
}

extern "C" void kernel_launch(void* const* d_in, const int* in_sizes, int n_in,
                              void* d_out, int out_size, void* d_ws, size_t ws_size,
                              hipStream_t stream) {
  const float* hidden = (const float*)d_in[0];
  const float* Wih    = (const float*)d_in[1];
  const float* Whh    = (const float*)d_in[2];
  const float* bih    = (const float*)d_in[3];
  const float* bhh    = (const float*)d_in[4];
  const float* Wo     = (const float*)d_in[5];
  const float* bo     = (const float*)d_in[6];
  float* out = (float*)d_out;

  uint16_t* wbase = (uint16_t*)d_ws;
  float* bb = (float*)((char*)d_ws + BIAS_BYTE_OFF);   // needs ~0.94 MB of ws

  prep_weights<<<(TOT_W_ELEMS + 255) / 256, 256, 0, stream>>>(Wih, Whh, Wo, wbase);
  prep_bias<<<(1056 + 255) / 256, 256, 0, stream>>>(bih, bhh, bo, bb);
  gru_decoder_main<<<BATCH / 32, 32, 0, stream>>>(hidden, wbase, bb, out);
}
